// PositionEncoder_82609400971881
// MI455X (gfx1250) — matrix operand order repaired and checked by reading, not compiled, not run on hardware
//
#include <hip/hip_runtime.h>
#include <math.h>

typedef __attribute__((ext_vector_type(16))) _Float16 v16h;
typedef __attribute__((ext_vector_type(8)))  float    v8f;
typedef _Float16 half_t;

#define N_NODES 4096
#define N_EDGES 8192
#define M_MOLS  128
#define NDIM    128
#define EDIM    64
#define PQ      16
#define LAYERS  4
#define TAU     0.25f
#define FRICTION 0.1f

__device__ __forceinline__ float sigmoidf_(float x) { return 1.0f / (1.0f + expf(-x)); }
__device__ __forceinline__ float softplusf_(float x) { return (x > 20.0f) ? x : log1pf(expf(x)); }

// ---------------------------------------------------------------------------
// Fused node-weight matrix value: W[k, c] for k in [0,128), c in [0,32):
//   cols 0..15 = Wp, 16 = We[0:128], 17 = We[192:320], 18 = Wt[0:128],
//   19 = Wu[0:128], 20..31 = 0
// ---------------------------------------------------------------------------
__device__ __forceinline__ float wcomb_val(const float* __restrict__ Wp,
                                           const float* __restrict__ We,
                                           const float* __restrict__ Wt,
                                           const float* __restrict__ Wu,
                                           int k, int c) {
  if (c < 16)  return Wp[k * PQ + c];
  if (c == 16) return We[k];
  if (c == 17) return We[192 + k];
  if (c == 18) return Wt[k];
  if (c == 19) return Wu[k];
  return 0.0f;
}

// ---------------------------------------------------------------------------
// Setup: pre-swizzle B matrices into per-lane WMMA fragment layout
// (v_wmma_f32_16x16x32_f16 B operand: lane = col N (0..15 / repeat), halves
//  0..7 = K kbase..+7, 8..15 = K kbase+16..+23, where kbase = kstep*32 + (lane<16 ? 0 : 8)).
// Fragment f stored at Wswz[((f*32 + lane)*16 + h)]  -> 32 contiguous bytes
// per lane, loadable as one v16h.
//   WcombSwz: f = kstep*2 + coltile   (8 fragments, 4096 halves)
//   WqSwz:    f = kstep               (4 fragments, 2048 halves)
// Also computes per-node molecule ids.
// ---------------------------------------------------------------------------
__global__ void k_setup(const float* __restrict__ Wp, const float* __restrict__ We,
                        const float* __restrict__ Wt, const float* __restrict__ Wu,
                        const float* __restrict__ Wq, const float* __restrict__ mnm,
                        half_t* __restrict__ WcombSwz, half_t* __restrict__ WqSwz,
                        int* __restrict__ mol_id) {
  int t = blockIdx.x * blockDim.x + threadIdx.x;

  if (t < 8 * 32 * 16) {  // WcombSwz entries
    int h    = t & 15;
    int lane = (t >> 4) & 31;
    int f    = t >> 9;
    int ks = f >> 1, ct = f & 1;
    int k   = ks * 32 + ((lane < 16) ? 0 : 8) + ((h < 8) ? h : h + 8);
    int col = (lane & 15) + ct * 16;
    WcombSwz[t] = (half_t)wcomb_val(Wp, We, Wt, Wu, k, col);
  }
  if (t < 4 * 32 * 16) {  // WqSwz entries
    int h    = t & 15;
    int lane = (t >> 4) & 31;
    int ks   = t >> 9;
    int k   = ks * 32 + ((lane < 16) ? 0 : 8) + ((h < 8) ? h : h + 8);
    int col = lane & 15;
    WqSwz[t] = (half_t)Wq[k * PQ + col];
  }
  if (t < N_NODES) {
    int id = 0;
    for (int m = 0; m < M_MOLS; ++m)
      if (mnm[(size_t)m * N_NODES + t] != 0.0f) { id = m; break; }
    mol_id[t] = id;
  }
}

// ---------------------------------------------------------------------------
// WMMA A loader (f32 source, convert to f16):
// lane holds row M = lane%16; halves 0..7 = K kbase..+7, 8..15 = K kbase+16..+23
// with kbase = kstep*32 + (lane<16 ? 0 : 8).
// ---------------------------------------------------------------------------
__device__ __forceinline__ v16h load_a_f32(const float* __restrict__ A, int row, int kbase) {
  v16h a;
  const float* p = A + (size_t)row * NDIM + kbase;
#pragma unroll
  for (int h = 0; h < 8; ++h) { a[h] = (half_t)p[h]; a[8 + h] = (half_t)p[16 + h]; }
  return a;
}

// ---------------------------------------------------------------------------
// Node GEMM: [4096,128](f32) x [128,32](f16) via v_wmma_f32_16x16x32_f16.
// Writes p0 = tanh(.+bp), sum_p init, and per-node scalars a_eu,a_ev,a_t,a_u.
// 8 waves/block, 16 rows/wave -> 32 blocks.
// ---------------------------------------------------------------------------
__global__ void k_node_mm(const float* __restrict__ vfeat, const half_t* __restrict__ WcombSwz,
                          const float* __restrict__ bp,
                          float* __restrict__ p0, float* __restrict__ sum_p,
                          float* __restrict__ a_eu, float* __restrict__ a_ev,
                          float* __restrict__ a_t, float* __restrict__ a_u) {
  int wave = threadIdx.x >> 5, lane = threadIdx.x & 31;
  int rowBase = (blockIdx.x * 8 + wave) * 16;
  int col = lane & 15;
  int hi  = lane >> 4;

  v8f c0 = {}; v8f c1 = {};
#pragma unroll
  for (int ks = 0; ks < 4; ++ks) {
    v16h a  = load_a_f32(vfeat, rowBase + col, ks * 32 + hi * 8);
    v16h b0 = *(const v16h*)(WcombSwz + (((ks * 2 + 0) * 32 + lane) << 4));
    v16h b1 = *(const v16h*)(WcombSwz + (((ks * 2 + 1) * 32 + lane) << 4));
    c0 = __builtin_amdgcn_wmma_f32_16x16x32_f16(false, a, false, b0, (short)0, c0, false, false);
    c1 = __builtin_amdgcn_wmma_f32_16x16x32_f16(false, a, false, b1, (short)0, c1, false, false);
  }

  float bpc = bp[col];
#pragma unroll
  for (int r = 0; r < 8; ++r) {
    int row = rowBase + hi * 8 + r;
    float pv = tanhf(c0[r] + bpc);
    p0[row * PQ + col]    = pv;
    sum_p[row * PQ + col] = pv;
    float s = c1[r];
    if (col == 0)      a_eu[row] = s;
    else if (col == 1) a_ev[row] = s;
    else if (col == 2) a_t[row]  = s;
    else if (col == 3) a_u[row]  = s;
  }
}

// ---------------------------------------------------------------------------
// Q GEMM: [4096,128](f32 = e@v) x [128,16](f16). q0 = tanh(.+bq), sum_q init.
// ---------------------------------------------------------------------------
__global__ void k_q_mm(const float* __restrict__ ev, const half_t* __restrict__ WqSwz,
                       const float* __restrict__ bq,
                       float* __restrict__ q0, float* __restrict__ sum_q) {
  int wave = threadIdx.x >> 5, lane = threadIdx.x & 31;
  int rowBase = (blockIdx.x * 8 + wave) * 16;
  int col = lane & 15;
  int hi  = lane >> 4;

  v8f c0 = {};
#pragma unroll
  for (int ks = 0; ks < 4; ++ks) {
    v16h a = load_a_f32(ev, rowBase + col, ks * 32 + hi * 8);
    v16h b = *(const v16h*)(WqSwz + ((ks * 32 + lane) << 4));
    c0 = __builtin_amdgcn_wmma_f32_16x16x32_f16(false, a, false, b, (short)0, c0, false, false);
  }

  float bqc = bq[col];
#pragma unroll
  for (int r = 0; r < 8; ++r) {
    int row = rowBase + hi * 8 + r;
    float qv = tanhf(c0[r] + bqc);
    q0[row * PQ + col]    = qv;
    sum_q[row * PQ + col] = qv;
  }
}

// ---------------------------------------------------------------------------
// Edge gate: w[j] = sigmoid(a_eu[u] + e_f[j].We[128:192] + a_ev[v] + be)
// ---------------------------------------------------------------------------
__global__ void k_edge_w(const float* __restrict__ efeat, const float* __restrict__ We,
                         const float* __restrict__ be,
                         const int* __restrict__ us, const int* __restrict__ vs,
                         const float* __restrict__ a_eu, const float* __restrict__ a_ev,
                         float* __restrict__ w) {
  __shared__ float Wel[EDIM];
  if (threadIdx.x < EDIM) Wel[threadIdx.x] = We[NDIM + threadIdx.x];
  __syncthreads();
  int j = blockIdx.x * blockDim.x + threadIdx.x;
  if (j >= N_EDGES) return;
  const float* er = efeat + (size_t)j * EDIM;
  float dot = 0.0f;
#pragma unroll 8
  for (int k = 0; k < EDIM; ++k) dot += er[k] * Wel[k];
  float z = a_eu[us[j]] + dot + a_ev[vs[j]] + be[0];
  w[j] = sigmoidf_(z);
}

// ---------------------------------------------------------------------------
// ev = e @ v  (sparse): edge j adds w_j*(v[u]+v[v]) to rows u and v.
// One wave per edge; lane covers 4 dims (float4 gather + f32 atomics).
// ---------------------------------------------------------------------------
__global__ void k_ev_scatter(const float* __restrict__ vfeat,
                             const int* __restrict__ us, const int* __restrict__ vs,
                             const float* __restrict__ w, float* __restrict__ ev) {
  int gid  = blockIdx.x * blockDim.x + threadIdx.x;
  int edge = gid >> 5;
  int lane = gid & 31;
  if (edge >= N_EDGES) return;
  int u = us[edge], v = vs[edge];
  float wj = w[edge];
  int k0 = lane * 4;
  float4 xu = *(const float4*)(vfeat + (size_t)u * NDIM + k0);
  if (u == v) {
    atomicAdd(&ev[(size_t)u * NDIM + k0 + 0], wj * xu.x);
    atomicAdd(&ev[(size_t)u * NDIM + k0 + 1], wj * xu.y);
    atomicAdd(&ev[(size_t)u * NDIM + k0 + 2], wj * xu.z);
    atomicAdd(&ev[(size_t)u * NDIM + k0 + 3], wj * xu.w);
  } else {
    float4 xv = *(const float4*)(vfeat + (size_t)v * NDIM + k0);
    float y0 = wj * (xu.x + xv.x), y1 = wj * (xu.y + xv.y);
    float y2 = wj * (xu.z + xv.z), y3 = wj * (xu.w + xv.w);
    atomicAdd(&ev[(size_t)u * NDIM + k0 + 0], y0);
    atomicAdd(&ev[(size_t)u * NDIM + k0 + 1], y1);
    atomicAdd(&ev[(size_t)u * NDIM + k0 + 2], y2);
    atomicAdd(&ev[(size_t)u * NDIM + k0 + 3], y3);
    atomicAdd(&ev[(size_t)v * NDIM + k0 + 0], y0);
    atomicAdd(&ev[(size_t)v * NDIM + k0 + 1], y1);
    atomicAdd(&ev[(size_t)v * NDIM + k0 + 2], y2);
    atomicAdd(&ev[(size_t)v * NDIM + k0 + 3], y3);
  }
}

// ---------------------------------------------------------------------------
// eq = e @ (q . WuP)  (scalar sparse matvec, r computed inline per endpoint)
// ---------------------------------------------------------------------------
__global__ void k_eq_scatter(const float* __restrict__ q, const float* __restrict__ Wu,
                             const int* __restrict__ us, const int* __restrict__ vs,
                             const float* __restrict__ w, float* __restrict__ eq) {
  __shared__ float WuP[PQ];
  if (threadIdx.x < PQ) WuP[threadIdx.x] = Wu[NDIM + threadIdx.x];
  __syncthreads();
  int j = blockIdx.x * blockDim.x + threadIdx.x;
  if (j >= N_EDGES) return;
  int u = us[j], v = vs[j];
  float wj = w[j];
  float ru = 0.0f;
#pragma unroll
  for (int k = 0; k < PQ; ++k) ru += q[(size_t)u * PQ + k] * WuP[k];
  if (u == v) {
    atomicAdd(&eq[u], wj * ru);
  } else {
    float rv = 0.0f;
#pragma unroll
    for (int k = 0; k < PQ; ++k) rv += q[(size_t)v * PQ + k] * WuP[k];
    float y = wj * (ru + rv);
    atomicAdd(&eq[u], y);
    atomicAdd(&eq[v], y);
  }
}

// ---------------------------------------------------------------------------
// es = e @ sigmoid(a_u + eq)   (su computed inline per endpoint)
// ---------------------------------------------------------------------------
__global__ void k_es_scatter(const float* __restrict__ a_u, const float* __restrict__ eq,
                             const int* __restrict__ us, const int* __restrict__ vs,
                             const float* __restrict__ w, float* __restrict__ es) {
  int j = blockIdx.x * blockDim.x + threadIdx.x;
  if (j >= N_EDGES) return;
  int u = us[j], v = vs[j];
  float wj = w[j];
  float su = sigmoidf_(a_u[u] + eq[u]);
  if (u == v) {
    atomicAdd(&es[u], wj * su);
  } else {
    float sv = sigmoidf_(a_u[v] + eq[v]);
    float y = wj * (su + sv);
    atomicAdd(&es[u], y);
    atomicAdd(&es[v], y);
  }
}

// ---------------------------------------------------------------------------
// Per-layer node update:
//   zt = a_t + p.WtP, st = sigmoid(zt); gp = st*WtP
//   zu = a_u + eq;                       gq = es*WuP
//   dq = gp; dp = -gq - FRICTION*gp
//   p' = p + TAU*dp; q' = q + TAU*dq; running sums; losses; h/d on last layer
// ---------------------------------------------------------------------------
__global__ void k_update(const float* __restrict__ p_cur, const float* __restrict__ q_cur,
                         float* __restrict__ p_next, float* __restrict__ q_next,
                         float* __restrict__ sum_p, float* __restrict__ sum_q,
                         const float* __restrict__ a_t, const float* __restrict__ a_u,
                         const float* __restrict__ eq, const float* __restrict__ es,
                         const float* __restrict__ Wt, const float* __restrict__ Wu,
                         const int* __restrict__ mol_id,
                         float* __restrict__ sloss, float* __restrict__ cmol,
                         float* __restrict__ hmol, float* __restrict__ dmol, int layer) {
  __shared__ float WtP[PQ], WuP[PQ];
  if (threadIdx.x < PQ) {
    WtP[threadIdx.x] = Wt[NDIM + threadIdx.x];
    WuP[threadIdx.x] = Wu[NDIM + threadIdx.x];
  }
  __syncthreads();
  int n = blockIdx.x * blockDim.x + threadIdx.x;
  if (n >= N_NODES) return;

  float zt = a_t[n];
#pragma unroll
  for (int k = 0; k < PQ; ++k) zt += p_cur[(size_t)n * PQ + k] * WtP[k];
  float st  = sigmoidf_(zt);
  float zu  = a_u[n] + eq[n];
  float esn = es[n];
  int   mid = mol_id[n];

  float sl = 0.0f;
#pragma unroll
  for (int k = 0; k < PQ; ++k) {
    float gp = st * WtP[k];
    float gq = esn * WuP[k];
    float dp = -gq - FRICTION * gp;
    float dq = gp;
    float pc = p_cur[(size_t)n * PQ + k];
    float pn = pc + TAU * dp;
    float qn = q_cur[(size_t)n * PQ + k] + TAU * dq;
    p_next[(size_t)n * PQ + k] = pn;
    q_next[(size_t)n * PQ + k] = qn;
    sum_p[(size_t)n * PQ + k] += pn;   // unique (n,k) per thread
    sum_q[(size_t)n * PQ + k] += qn;
    float diff = dq - pc;
    sl += diff * diff;
    atomicAdd(&cmol[(size_t)layer * M_MOLS * PQ + (size_t)mid * PQ + k], dp);
  }
  atomicAdd(&sloss[layer], sl);

  if (layer == LAYERS - 1) {
    float tn = softplusf_(zt);
    float un = softplusf_(zu);
    atomicAdd(&hmol[mid], tn + un);
    float w2 = 0.0f;
#pragma unroll
    for (int k = 0; k < PQ; ++k) w2 += WtP[k] * WtP[k];
    atomicAdd(&dmol[mid], st * st * w2);
  }
}

// ---------------------------------------------------------------------------
// Finalize: averages -> out[0:131072]
// ---------------------------------------------------------------------------
__global__ void k_avg(const float* __restrict__ sum_p, const float* __restrict__ sum_q,
                      float* __restrict__ out) {
  int t = blockIdx.x * blockDim.x + threadIdx.x;
  const float inv = 1.0f / (float)(LAYERS + 1);
  const int NP = N_NODES * PQ;
  if (t < NP)          out[t] = sum_p[t] * inv;
  else if (t < 2 * NP) out[t] = sum_q[t - NP] * inv;
}

// Scalars + per-molecule outputs -> out[131072 : 131330]
__global__ void k_final(const float* __restrict__ sloss, const float* __restrict__ cmol,
                        const float* __restrict__ hmol, const float* __restrict__ dmol,
                        float* __restrict__ out) {
  float* base = out + 2 * N_NODES * PQ;
  int t = threadIdx.x;
  if (t == 0) {
    float s = 0.0f;
    for (int i = 0; i < LAYERS; ++i) s += sqrtf(sloss[i]);
    base[0] = s;
    float c = 0.0f;
    for (int i = 0; i < LAYERS; ++i) {
      float acc = 0.0f;
      for (int j = 0; j < M_MOLS * PQ; ++j) {
        float x = cmol[(size_t)i * M_MOLS * PQ + j];
        acc += x * x;
      }
      c += TAU * sqrtf(acc);
    }
    base[1] = c;
  }
  if (t < M_MOLS) {
    base[2 + t]          = hmol[t];
    base[2 + M_MOLS + t] = dmol[t];
  }
}

// ---------------------------------------------------------------------------
extern "C" void kernel_launch(void* const* d_in, const int* in_sizes, int n_in,
                              void* d_out, int out_size, void* d_ws, size_t ws_size,
                              hipStream_t stream) {
  const float* vfeat = (const float*)d_in[0];
  const float* efeat = (const float*)d_in[1];
  const int*   us    = (const int*)d_in[2];
  const int*   vs    = (const int*)d_in[3];
  const float* mnm   = (const float*)d_in[4];
  // d_in[5] mol_node_mask, d_in[6] node_edge_matrix, d_in[7] node_edge_mask: unused
  const float* We = (const float*)d_in[8];
  const float* be = (const float*)d_in[9];
  const float* Wp = (const float*)d_in[10];
  const float* bp = (const float*)d_in[11];
  const float* Wq = (const float*)d_in[12];
  const float* bq = (const float*)d_in[13];
  const float* Wt = (const float*)d_in[14];
  const float* Wu = (const float*)d_in[15];
  float* out = (float*)d_out;

  char* ws = (char*)d_ws;
  size_t off = 0;
  auto alloc = [&](size_t bytes) -> char* {
    char* p = ws + off;
    off += (bytes + 255) & ~(size_t)255;
    return p;
  };

  half_t* WcombSwz = (half_t*)alloc(8 * 32 * 16 * sizeof(half_t));
  half_t* WqSwz    = (half_t*)alloc(4 * 32 * 16 * sizeof(half_t));
  float* a_eu = (float*)alloc(N_NODES * sizeof(float));
  float* a_ev = (float*)alloc(N_NODES * sizeof(float));
  float* a_t  = (float*)alloc(N_NODES * sizeof(float));
  float* a_u  = (float*)alloc(N_NODES * sizeof(float));
  float* w    = (float*)alloc(N_EDGES * sizeof(float));
  float* ev   = (float*)alloc((size_t)N_NODES * NDIM * sizeof(float));
  float* pbuf0 = (float*)alloc((size_t)N_NODES * PQ * sizeof(float));
  float* pbuf1 = (float*)alloc((size_t)N_NODES * PQ * sizeof(float));
  float* qbuf0 = (float*)alloc((size_t)N_NODES * PQ * sizeof(float));
  float* qbuf1 = (float*)alloc((size_t)N_NODES * PQ * sizeof(float));
  float* sum_p = (float*)alloc((size_t)N_NODES * PQ * sizeof(float));
  float* sum_q = (float*)alloc((size_t)N_NODES * PQ * sizeof(float));
  float* eq    = (float*)alloc(N_NODES * sizeof(float));
  float* es    = (float*)alloc(N_NODES * sizeof(float));
  float* sloss = (float*)alloc(LAYERS * sizeof(float));
  float* cmol  = (float*)alloc((size_t)LAYERS * M_MOLS * PQ * sizeof(float));
  float* hmol  = (float*)alloc(M_MOLS * sizeof(float));
  float* dmol  = (float*)alloc(M_MOLS * sizeof(float));
  int*   mol_id = (int*)alloc(N_NODES * sizeof(int));

  // zero accumulators (capture-safe async memsets)
  hipMemsetAsync(ev, 0, (size_t)N_NODES * NDIM * sizeof(float), stream);
  hipMemsetAsync(sloss, 0, LAYERS * sizeof(float), stream);
  hipMemsetAsync(cmol, 0, (size_t)LAYERS * M_MOLS * PQ * sizeof(float), stream);
  hipMemsetAsync(hmol, 0, M_MOLS * sizeof(float), stream);
  hipMemsetAsync(dmol, 0, M_MOLS * sizeof(float), stream);

  k_setup<<<16, 256, 0, stream>>>(Wp, We, Wt, Wu, Wq, mnm, WcombSwz, WqSwz, mol_id);

  // p0 + per-node scalars (WMMA)
  k_node_mm<<<32, 256, 0, stream>>>(vfeat, WcombSwz, bp, pbuf0, sum_p, a_eu, a_ev, a_t, a_u);

  // edge gates
  k_edge_w<<<32, 256, 0, stream>>>(efeat, We, be, us, vs, a_eu, a_ev, w);

  // ev = e @ v (wave per edge)
  k_ev_scatter<<<(N_EDGES * 32) / 256, 256, 0, stream>>>(vfeat, us, vs, w, ev);

  // q0 (WMMA)
  k_q_mm<<<32, 256, 0, stream>>>(ev, WqSwz, bq, qbuf0, sum_q);

  float* pc = pbuf0; float* pn = pbuf1;
  float* qc = qbuf0; float* qn = qbuf1;
  for (int layer = 0; layer < LAYERS; ++layer) {
    hipMemsetAsync(eq, 0, N_NODES * sizeof(float), stream);
    hipMemsetAsync(es, 0, N_NODES * sizeof(float), stream);
    k_eq_scatter<<<32, 256, 0, stream>>>(qc, Wu, us, vs, w, eq);
    k_es_scatter<<<32, 256, 0, stream>>>(a_u, eq, us, vs, w, es);
    k_update<<<16, 256, 0, stream>>>(pc, qc, pn, qn, sum_p, sum_q, a_t, a_u, eq, es,
                                     Wt, Wu, mol_id, sloss, cmol, hmol, dmol, layer);
    float* tp = pc; pc = pn; pn = tp;
    float* tq = qc; qc = qn; qn = tq;
  }

  k_avg<<<(2 * N_NODES * PQ) / 256, 256, 0, stream>>>(sum_p, sum_q, out);
  k_final<<<1, 256, 0, stream>>>(sloss, cmol, hmol, dmol, out);
}
